// DeepseekV3MTPSpeculator_27032524161333
// MI455X (gfx1250) — matrix operand order repaired and checked by reading, not compiled, not run on hardware
//
#include <hip/hip_runtime.h>

// ---------------- model dims ----------------
#define TT      1024
#define HD      2048
#define NEXP    8
#define MOEI    1024
#define NH      16
#define QLORA   512
#define KVLORA  256
#define NOPE    64
#define ROPED   32
#define VD      64
#define QKD     96      // NOPE + ROPE
#define QDIM    1536    // NH*QKD
#define KVDIM   2048    // NH*(NOPE+VD)
#define KVA     288     // KVLORA + ROPE
#define NEGBIG  -1e30f

typedef __attribute__((ext_vector_type(16))) __bf16 v16bf;
typedef __attribute__((ext_vector_type(8)))  float  v8f;

union Frag { unsigned int u[8]; uint4 q[2]; v16bf v; };

static __device__ __forceinline__ unsigned short f2bf(float f) {
    unsigned int u = __float_as_uint(f);
    unsigned int r = ((u >> 16) & 1u) + 0x7fffu;   // round-to-nearest-even
    return (unsigned short)((u + r) >> 16);
}
static __device__ __forceinline__ float bf2f(unsigned short h) {
    return __uint_as_float(((unsigned int)h) << 16);
}

// block-wide sum over 256 threads (8 waves, wave32)
static __device__ __forceinline__ float blk_sum(float s, float* red) {
    int lane = threadIdx.x & 31, wv = threadIdx.x >> 5;
#pragma unroll
    for (int m = 16; m; m >>= 1) s += __shfl_xor(s, m, 32);
    if (lane == 0) red[wv] = s;
    __syncthreads();
    if (threadIdx.x == 0) { float t = 0.f; for (int i = 0; i < 8; i++) t += red[i]; red[8] = t; }
    __syncthreads();
    return red[8];
}

// ---------------- embed + rmsnorm + concat -> bf16 ----------------
__global__ __launch_bounds__(256) void embed_cat_kernel(
    const int* __restrict__ ids, const float* __restrict__ embed,
    const float* __restrict__ prev, const float* __restrict__ enw,
    const float* __restrict__ hnw, unsigned short* __restrict__ cat)
{
    __shared__ float red[9];
    int t = blockIdx.x;
    const float* e = embed + (size_t)ids[t] * HD;
    const float* p = prev  + (size_t)t * HD;
    float s1 = 0.f, s2 = 0.f;
    for (int i = threadIdx.x; i < HD; i += 256) { float a = e[i]; s1 += a * a; }
    for (int i = threadIdx.x; i < HD; i += 256) { float b = p[i]; s2 += b * b; }
    float tot1 = blk_sum(s1, red);
    float inv1 = rsqrtf(tot1 / HD + 1e-6f);
    float tot2 = blk_sum(s2, red);
    float inv2 = rsqrtf(tot2 / HD + 1e-6f);
    unsigned short* c0 = cat + (size_t)t * (2 * HD);
    for (int i = threadIdx.x; i < HD; i += 256) {
        c0[i]      = f2bf(e[i] * inv1 * enw[i]);
        c0[HD + i] = f2bf(p[i] * inv2 * hnw[i]);
    }
}

// ---------------- rmsnorm f32 -> bf16 ----------------
__global__ __launch_bounds__(256) void rmsnorm_kernel(
    const float* __restrict__ in, int istride, int L,
    const float* __restrict__ w, unsigned short* __restrict__ out, int ostride)
{
    __shared__ float red[9];
    int t = blockIdx.x;
    const float* x = in + (size_t)t * istride;
    float s = 0.f;
    for (int i = threadIdx.x; i < L; i += 256) { float v = x[i]; s += v * v; }
    float tot = blk_sum(s, red);
    float inv = rsqrtf(tot / L + 1e-6f);
    for (int i = threadIdx.x; i < L; i += 256)
        out[(size_t)t * ostride + i] = f2bf(x[i] * inv * w[i]);
}

// ---------------- generic bf16 WMMA GEMM, 128x128 tile, BK=32, double-buffered ----------------
// modes: 0: outf=acc  1: outf=acc+aux1  3: outbf=silu(acc)*aux1
//        4: outf[sidx[row]] += rowscale[row]*acc   5: outf=acc+aux1+aux2
__global__ __launch_bounds__(256) void gemm_bf16_wmma(
    const unsigned short* __restrict__ A, int lda,
    const float* __restrict__ B, int ldb, int Ncols, int K, int M,
    const int* __restrict__ gidx, const int* __restrict__ cntp,
    int mode,
    float* __restrict__ outf, int ldo,
    unsigned short* __restrict__ outbf, int ldob,
    const float* __restrict__ aux1, int lda1,
    const float* __restrict__ aux2,
    const float* __restrict__ rowscale, const int* __restrict__ sidx)
{
    int tn = blockIdx.x, tm = blockIdx.y;
    int Meff = cntp ? *cntp : M;
    if (tm * 128 >= Meff) return;

    __shared__ unsigned short As[2][128][48];   // row-major, padded (96B stride)
    __shared__ unsigned short Bt[2][128][48];   // B transposed: Bt[n][k]

    int tid = threadIdx.x, wave = tid >> 5, lane = tid & 31;
    int lr = lane & 15, hi = lane >> 4;

    v8f acc[8];
    v8f zero = {0.f, 0.f, 0.f, 0.f, 0.f, 0.f, 0.f, 0.f};
#pragma unroll
    for (int nt = 0; nt < 8; nt++) acc[nt] = zero;

    // ---- hoisted A-stage addressing (gather index loaded ONCE) ----
    int ar_ = tid >> 1, acs = (tid & 1) * 16;
    int aslot = tm * 128 + ar_;
    const unsigned short* aptr = nullptr;
    if (aslot < Meff) {
        int arow = gidx ? gidx[aslot] : aslot;
        aptr = A + (size_t)arow * lda + acs;
    }
    // ---- hoisted B-stage addressing + bounds ----
    int bk = tid >> 3, bn0 = (tid & 7) * 16;
    const float* bbase = B + (size_t)bk * ldb + tn * 128 + bn0;
    bool binb[4];
#pragma unroll
    for (int j4 = 0; j4 < 4; j4++) binb[j4] = (tn * 128 + bn0 + j4 * 4 + 4) <= Ncols;

    int nkt = K >> 5;

    // register staging for the software pipeline
    uint4 a0 = {0, 0, 0, 0}, a1 = {0, 0, 0, 0};
    float4 bv[4];

    auto load_tile = [&](int kt) {
        if (kt + 2 < nkt && (tid & 7) == 0)
            __builtin_prefetch(bbase + (size_t)(kt + 2) * 32 * ldb, 0, 1);
        uint4 z = {0, 0, 0, 0};
        a0 = z; a1 = z;
        if (aptr) {
            a0 = *(const uint4*)(aptr + kt * 32);
            a1 = *(const uint4*)(aptr + kt * 32 + 8);
        }
        const float* bp = bbase + (size_t)kt * 32 * ldb;
#pragma unroll
        for (int j4 = 0; j4 < 4; j4++) {
            float4 v = {0.f, 0.f, 0.f, 0.f};
            if (binb[j4]) v = *(const float4*)(bp + j4 * 4);
            bv[j4] = v;
        }
    };
    auto store_tile = [&](int buf) {
        *(uint4*)&As[buf][ar_][acs]     = a0;
        *(uint4*)&As[buf][ar_][acs + 8] = a1;
#pragma unroll
        for (int j4 = 0; j4 < 4; j4++) {
            Bt[buf][bn0 + j4 * 4 + 0][bk] = f2bf(bv[j4].x);
            Bt[buf][bn0 + j4 * 4 + 1][bk] = f2bf(bv[j4].y);
            Bt[buf][bn0 + j4 * 4 + 2][bk] = f2bf(bv[j4].z);
            Bt[buf][bn0 + j4 * 4 + 3][bk] = f2bf(bv[j4].w);
        }
    };
    auto compute = [&](int buf) {
        Frag af; // A fragment 16x32: lane=row(lr), K = hi*16..+15
        int ar = wave * 16 + lr;
        af.q[0] = *(const uint4*)&As[buf][ar][hi * 16];
        af.q[1] = *(const uint4*)&As[buf][ar][hi * 16 + 8];
        Frag bf8[8]; // preload ALL B fragments, then chain the 8 WMMAs
#pragma unroll
        for (int nt = 0; nt < 8; nt++) {
            bf8[nt].q[0] = *(const uint4*)&Bt[buf][nt * 16 + lr][hi * 16];
            bf8[nt].q[1] = *(const uint4*)&Bt[buf][nt * 16 + lr][hi * 16 + 8];
        }
#pragma unroll
        for (int nt = 0; nt < 8; nt++)
            acc[nt] = __builtin_amdgcn_wmma_f32_16x16x32_bf16(
                false, af.v, false, bf8[nt].v, (short)0, acc[nt], false, false);
    };

    // software pipeline: global->regs (kt+1) overlaps LDS->WMMA (kt); 1 barrier/tile
    load_tile(0);
    store_tile(0);
    __syncthreads();
    for (int kt = 0; kt < nkt; kt++) {
        if (kt + 1 < nkt) load_tile(kt + 1);
        compute(kt & 1);
        if (kt + 1 < nkt) {
            store_tile((kt + 1) & 1);
            __syncthreads();
        }
    }

    // epilogue: acc[nt][i] -> row = tm*128 + wave*16 + hi*8 + i, col = tn*128 + nt*16 + lr
#pragma unroll
    for (int i = 0; i < 8; i++) {
        int slot = tm * 128 + wave * 16 + hi * 8 + i;
        if (slot >= Meff) continue;
#pragma unroll
        for (int nt = 0; nt < 8; nt++) {
            int col = tn * 128 + nt * 16 + lr;
            if (col >= Ncols) continue;
            float v = acc[nt][i];
            if (mode == 0) {
                outf[(size_t)slot * ldo + col] = v;
            } else if (mode == 1) {
                outf[(size_t)slot * ldo + col] = v + aux1[(size_t)slot * lda1 + col];
            } else if (mode == 3) {
                float u = aux1[(size_t)slot * lda1 + col];
                float sg = v / (1.f + __expf(-v));
                outbf[(size_t)slot * ldob + col] = f2bf(sg * u);
            } else if (mode == 4) {
                int trow = sidx[slot];
                outf[(size_t)trow * ldo + col] += rowscale[slot] * v;
            } else { // 5
                outf[(size_t)slot * ldo + col] =
                    v + aux1[(size_t)slot * lda1 + col] + aux2[(size_t)slot * lda1 + col];
            }
        }
    }
}

// ---------------- RoPE + QKV assembly: q f32 -> qf bf16 (scaled), kv/kva -> kf bf16 ----------------
__global__ __launch_bounds__(128) void prep_qkv_kernel(
    const float* __restrict__ q, const float* __restrict__ kva,
    const float* __restrict__ kv, const int* __restrict__ pos,
    unsigned short* __restrict__ qf, unsigned short* __restrict__ kf)
{
    int t = blockIdx.x;
    float p = (float)pos[t];
    const float scale = rsqrtf((float)QKD);
    for (int idx = threadIdx.x; idx < NH * 48; idx += 128) {
        int h = idx / 48, pr = idx % 48;
        size_t qb = (size_t)t * QDIM + h * QKD;
        size_t kb = (size_t)t * QDIM + h * QKD;
        if (pr < 32) { // nope dims 2pr, 2pr+1
            int d = 2 * pr;
            qf[qb + d]     = f2bf(q[(size_t)t * QDIM + h * QKD + d] * scale);
            qf[qb + d + 1] = f2bf(q[(size_t)t * QDIM + h * QKD + d + 1] * scale);
            kf[kb + d]     = f2bf(kv[(size_t)t * KVDIM + h * 128 + d]);
            kf[kb + d + 1] = f2bf(kv[(size_t)t * KVDIM + h * 128 + d + 1]);
        } else { // rope pair j
            int j = pr - 32;
            float invf = __powf(10000.f, -(2.f * j) / (float)ROPED);
            float ang = p * invf;
            float c = __cosf(ang), s = __sinf(ang);
            float x1 = q[(size_t)t * QDIM + h * QKD + NOPE + 2 * j];
            float x2 = q[(size_t)t * QDIM + h * QKD + NOPE + 2 * j + 1];
            qf[qb + NOPE + 2 * j]     = f2bf((x1 * c - x2 * s) * scale);
            qf[qb + NOPE + 2 * j + 1] = f2bf((x1 * s + x2 * c) * scale);
            float k1 = kva[(size_t)t * KVA + KVLORA + 2 * j];
            float k2 = kva[(size_t)t * KVA + KVLORA + 2 * j + 1];
            kf[kb + NOPE + 2 * j]     = f2bf(k1 * c - k2 * s);
            kf[kb + NOPE + 2 * j + 1] = f2bf(k1 * s + k2 * c);
        }
    }
}

// ---------------- flash attention: per (qblock of 128, head); KB=64 keys/iter ----------------
__global__ __launch_bounds__(256) void attn_flash_kernel(
    const unsigned short* __restrict__ qf, const unsigned short* __restrict__ kf,
    const float* __restrict__ kv, unsigned short* __restrict__ obf)
{
    int qb = blockIdx.x, h = blockIdx.y;
    int tid = threadIdx.x, wave = tid >> 5, lane = tid & 31;
    int lr = lane & 15, hi = lane >> 4;

    __shared__ unsigned short Ks[64][104];      // K tile row-major [key][d], acts as Bt
    __shared__ unsigned short Vt[64][64];       // V transposed [vd][key], acts as Bt
    __shared__ unsigned short Ps[8][16][64];    // per-wave P staging

    // Q fragments straight from global (A layout)
    Frag qa3[3];
    const unsigned short* qrow = qf + (size_t)(qb * 128 + wave * 16 + lr) * QDIM + h * QKD;
#pragma unroll
    for (int kc = 0; kc < 3; kc++) {
        qa3[kc].q[0] = *(const uint4*)(qrow + kc * 32 + hi * 16);
        qa3[kc].q[1] = *(const uint4*)(qrow + kc * 32 + hi * 16 + 8);
    }

    v8f zero = {0.f, 0.f, 0.f, 0.f, 0.f, 0.f, 0.f, 0.f};
    v8f O[4]; // 16 rows x 64 vdims per wave
#pragma unroll
    for (int vt = 0; vt < 4; vt++) O[vt] = zero;
    float m[8], l[8];
#pragma unroll
    for (int i = 0; i < 8; i++) { m[i] = NEGBIG; l[i] = 0.f; }

    int cmax = 2 * qb + 1;
    for (int c = 0; c <= cmax; c++) {
        __syncthreads();
        // stage K tile (bf16, direct) and V tile (f32 -> bf16, transposed)
        for (int i = tid; i < 64 * 12; i += 256) {
            int r = i / 12, c8 = (i % 12) * 8;
            *(uint4*)&Ks[r][c8] =
                *(const uint4*)(kf + (size_t)(c * 64 + r) * QDIM + h * QKD + c8);
        }
        for (int i = tid; i < 64 * 64; i += 256) {
            int key = i >> 6, vd = i & 63;
            Vt[vd][key] = f2bf(kv[(size_t)(c * 64 + key) * KVDIM + h * 128 + NOPE + vd]);
        }
        __syncthreads();

        // S = Q * K^T: per k-chunk preload 4 B fragments then chain 4 WMMAs
        v8f S[4];
#pragma unroll
        for (int nt = 0; nt < 4; nt++) S[nt] = zero;
#pragma unroll
        for (int kc = 0; kc < 3; kc++) {
            Frag kb4[4];
#pragma unroll
            for (int nt = 0; nt < 4; nt++) {
                kb4[nt].q[0] = *(const uint4*)&Ks[nt * 16 + lr][kc * 32 + hi * 16];
                kb4[nt].q[1] = *(const uint4*)&Ks[nt * 16 + lr][kc * 32 + hi * 16 + 8];
            }
#pragma unroll
            for (int nt = 0; nt < 4; nt++)
                S[nt] = __builtin_amdgcn_wmma_f32_16x16x32_bf16(
                    false, qa3[kc].v, false, kb4[nt].v, (short)0, S[nt], false, false);
        }
        // causal mask (only needed on the last two key blocks)
        if (c >= 2 * qb) {
#pragma unroll
            for (int nt = 0; nt < 4; nt++) {
                int col = c * 64 + nt * 16 + lr;
#pragma unroll
                for (int i = 0; i < 8; i++) {
                    int row = qb * 128 + wave * 16 + hi * 8 + i;
                    if (col > row) S[nt][i] = NEGBIG;
                }
            }
        }
        // online softmax
        float nm[8];
#pragma unroll
        for (int i = 0; i < 8; i++) {
            float p = S[0][i];
#pragma unroll
            for (int nt = 1; nt < 4; nt++) p = fmaxf(p, S[nt][i]);
#pragma unroll
            for (int mk = 1; mk < 16; mk <<= 1) p = fmaxf(p, __shfl_xor(p, mk, 32));
            nm[i] = fmaxf(m[i], p);
        }
#pragma unroll
        for (int i = 0; i < 8; i++) {
            float a = __expf(m[i] - nm[i]);
            l[i] *= a;
#pragma unroll
            for (int vt = 0; vt < 4; vt++) O[vt][i] *= a;
            m[i] = nm[i];
        }
        float ls[8];
#pragma unroll
        for (int i = 0; i < 8; i++) ls[i] = 0.f;
#pragma unroll
        for (int nt = 0; nt < 4; nt++)
#pragma unroll
            for (int i = 0; i < 8; i++) {
                float p = __expf(S[nt][i] - m[i]);
                S[nt][i] = p;
                ls[i] += p;
            }
#pragma unroll
        for (int i = 0; i < 8; i++) {
            float s = ls[i];
#pragma unroll
            for (int mk = 1; mk < 16; mk <<= 1) s += __shfl_xor(s, mk, 32);
            l[i] += s;
        }
        // stage P to LDS (wave-private) so it can be re-read as A fragments
#pragma unroll
        for (int nt = 0; nt < 4; nt++)
#pragma unroll
            for (int i = 0; i < 8; i++)
                Ps[wave][hi * 8 + i][nt * 16 + lr] = f2bf(S[nt][i]);
        __syncthreads();

        // O += P * V: preload fragments per k-chunk, then chain WMMAs
#pragma unroll
        for (int kc = 0; kc < 2; kc++) {
            Frag pa;
            pa.q[0] = *(const uint4*)&Ps[wave][lr][kc * 32 + hi * 16];
            pa.q[1] = *(const uint4*)&Ps[wave][lr][kc * 32 + hi * 16 + 8];
            Frag vb4[4];
#pragma unroll
            for (int vt = 0; vt < 4; vt++) {
                vb4[vt].q[0] = *(const uint4*)&Vt[vt * 16 + lr][kc * 32 + hi * 16];
                vb4[vt].q[1] = *(const uint4*)&Vt[vt * 16 + lr][kc * 32 + hi * 16 + 8];
            }
#pragma unroll
            for (int vt = 0; vt < 4; vt++)
                O[vt] = __builtin_amdgcn_wmma_f32_16x16x32_bf16(
                    false, pa.v, false, vb4[vt].v, (short)0, O[vt], false, false);
        }
    }

    // normalize and emit bf16 o
#pragma unroll
    for (int i = 0; i < 8; i++) {
        float inv = 1.f / (l[i] + 1e-30f);
        int row = qb * 128 + wave * 16 + hi * 8 + i;
#pragma unroll
        for (int vt = 0; vt < 4; vt++)
            obf[(size_t)row * (NH * VD) + h * VD + vt * 16 + lr] = f2bf(O[vt][i] * inv);
    }
}

// ---------------- MoE gate: sigmoid + top-2 ----------------
__global__ __launch_bounds__(256) void gate_kernel(
    const unsigned short* __restrict__ h2, const float* __restrict__ gw,
    const float* __restrict__ gb, int* __restrict__ sel, float* __restrict__ wts)
{
    __shared__ float lg[8];
    int t = blockIdx.x, wave = threadIdx.x >> 5, lane = threadIdx.x & 31;
    float s = 0.f;
    for (int k = lane; k < HD; k += 32)
        s += bf2f(h2[(size_t)t * HD + k]) * gw[(size_t)k * NEXP + wave];
#pragma unroll
    for (int mk = 16; mk; mk >>= 1) s += __shfl_xor(s, mk, 32);
    if (lane == 0) lg[wave] = s;
    __syncthreads();
    if (threadIdx.x == 0) {
        float sig[8], sc[8];
        for (int e = 0; e < NEXP; e++) {
            sig[e] = 1.f / (1.f + __expf(-lg[e]));
            sc[e] = sig[e] + gb[e];
        }
        int e0 = 0;
        for (int e = 1; e < NEXP; e++) if (sc[e] > sc[e0]) e0 = e;
        int e1 = -1;
        for (int e = 0; e < NEXP; e++) {
            if (e == e0) continue;
            if (e1 < 0 || sc[e] > sc[e1]) e1 = e;
        }
        float w0 = sig[e0], w1 = sig[e1];
        float den = w0 + w1 + 1e-20f;
        sel[t * 2] = e0; sel[t * 2 + 1] = e1;
        wts[t * 2] = (w0 / den) * 2.5f;      // fold ROUTED_SCALE
        wts[t * 2 + 1] = (w1 / den) * 2.5f;
    }
}

// deterministic (atomics-free) per-expert token lists
__global__ __launch_bounds__(256) void build_lists_kernel(
    const int* __restrict__ sel, const float* __restrict__ wts,
    int* __restrict__ idx, float* __restrict__ wslot, int* __restrict__ cnt)
{
    int e = threadIdx.x >> 5;
    if ((threadIdx.x & 31) != 0 || e >= NEXP) return;
    int c = 0;
    for (int t = 0; t < TT; t++) {
        if (sel[t * 2] == e)          { idx[e * TT + c] = t; wslot[e * TT + c] = wts[t * 2];     c++; }
        else if (sel[t * 2 + 1] == e) { idx[e * TT + c] = t; wslot[e * TT + c] = wts[t * 2 + 1]; c++; }
    }
    cnt[e] = c;
}

// ---------------- host orchestration ----------------
extern "C" void kernel_launch(void* const* d_in, const int* in_sizes, int n_in,
                              void* d_out, int out_size, void* d_ws, size_t ws_size,
                              hipStream_t stream) {
    (void)in_sizes; (void)n_in; (void)out_size; (void)ws_size;
    const int*   input_ids = (const int*)d_in[0];
    const int*   positions = (const int*)d_in[1];
    const float* prev      = (const float*)d_in[2];
    const float* embed     = (const float*)d_in[3];
    const float* enorm_w   = (const float*)d_in[4];
    const float* hnorm_w   = (const float*)d_in[5];
    const float* eh_proj_w = (const float*)d_in[6];
    const float* in_ln_w   = (const float*)d_in[7];
    const float* post_ln_w = (const float*)d_in[8];
    const float* q_a_w     = (const float*)d_in[9];
    const float* q_a_ln_w  = (const float*)d_in[10];
    const float* q_b_w     = (const float*)d_in[11];
    const float* kv_a_w    = (const float*)d_in[12];
    const float* kv_a_ln_w = (const float*)d_in[13];
    const float* kv_b_w    = (const float*)d_in[14];
    const float* o_w       = (const float*)d_in[15];
    const float* gate_w    = (const float*)d_in[16];
    const float* gate_bias = (const float*)d_in[17];
    const float* e_gate_w  = (const float*)d_in[18];
    const float* e_up_w    = (const float*)d_in[19];
    const float* e_down_w  = (const float*)d_in[20];
    const float* s_gate_w  = (const float*)d_in[21];
    const float* s_up_w    = (const float*)d_in[22];
    const float* s_down_w  = (const float*)d_in[23];
    float* out = (float*)d_out;

    char* base = (char*)d_ws;
    size_t off = 0;
    auto take = [&](size_t b) -> void* {
        void* p = base + off;
        off = (off + b + 255) & ~(size_t)255;
        return p;
    };
    unsigned short* cat_bf = (unsigned short*)take((size_t)TT * 2 * HD * 2);
    float*          x0     = (float*)take((size_t)TT * HD * 4);
    unsigned short* h_bf   = (unsigned short*)take((size_t)TT * HD * 2);
    float*          qa_f   = (float*)take((size_t)TT * QLORA * 4);
    unsigned short* qa_bf  = (unsigned short*)take((size_t)TT * QLORA * 2);
    float*          q_f    = (float*)take((size_t)TT * QDIM * 4);
    float*          kva_f  = (float*)take((size_t)TT * KVA * 4);
    unsigned short* kvc_bf = (unsigned short*)take((size_t)TT * KVLORA * 2);
    float*          kv_f   = (float*)take((size_t)TT * KVDIM * 4);
    unsigned short* qf_bf  = (unsigned short*)take((size_t)TT * QDIM * 2);
    unsigned short* kf_bf  = (unsigned short*)take((size_t)TT * QDIM * 2);
    unsigned short* o_bf   = (unsigned short*)take((size_t)TT * NH * VD * 2);
    float*          resid2 = (float*)take((size_t)TT * HD * 4);
    unsigned short* h2_bf  = (unsigned short*)take((size_t)TT * HD * 2);
    int*            sel    = (int*)take((size_t)TT * 2 * 4);
    float*          wts    = (float*)take((size_t)TT * 2 * 4);
    int*            eidx   = (int*)take((size_t)NEXP * TT * 4);
    float*          ew     = (float*)take((size_t)NEXP * TT * 4);
    int*            ecnt   = (int*)take(64);
    float*          u_f    = (float*)take((size_t)TT * MOEI * 4);
    unsigned short* a_bf   = (unsigned short*)take((size_t)TT * MOEI * 2);
    float*          routed = (float*)take((size_t)TT * HD * 4);

    dim3 blk(256);
    auto grid = [](int N) { return dim3((unsigned)((N + 127) / 128), 8); };

    // 1) embed + rmsnorm + concat
    embed_cat_kernel<<<TT, blk, 0, stream>>>(input_ids, embed, prev, enorm_w, hnorm_w, cat_bf);
    // 2) x0 = cat @ eh_proj   (1024 x 4096 x 2048)
    gemm_bf16_wmma<<<grid(HD), blk, 0, stream>>>(cat_bf, 2 * HD, eh_proj_w, HD, HD, 2 * HD, TT,
        nullptr, nullptr, 0, x0, HD, nullptr, 0, nullptr, 0, nullptr, nullptr, nullptr);
    // 3) h = rmsnorm(x0)
    rmsnorm_kernel<<<TT, blk, 0, stream>>>(x0, HD, HD, in_ln_w, h_bf, HD);
    // 4) q_a = h @ q_a_w ; rmsnorm
    gemm_bf16_wmma<<<grid(QLORA), blk, 0, stream>>>(h_bf, HD, q_a_w, QLORA, QLORA, HD, TT,
        nullptr, nullptr, 0, qa_f, QLORA, nullptr, 0, nullptr, 0, nullptr, nullptr, nullptr);
    rmsnorm_kernel<<<TT, blk, 0, stream>>>(qa_f, QLORA, QLORA, q_a_ln_w, qa_bf, QLORA);
    // 5) q = q_a @ q_b_w
    gemm_bf16_wmma<<<grid(QDIM), blk, 0, stream>>>(qa_bf, QLORA, q_b_w, QDIM, QDIM, QLORA, TT,
        nullptr, nullptr, 0, q_f, QDIM, nullptr, 0, nullptr, 0, nullptr, nullptr, nullptr);
    // 6) kv_a = h @ kv_a_w  (N=288, bounds-checked)
    gemm_bf16_wmma<<<grid(KVA), blk, 0, stream>>>(h_bf, HD, kv_a_w, KVA, KVA, HD, TT,
        nullptr, nullptr, 0, kva_f, KVA, nullptr, 0, nullptr, 0, nullptr, nullptr, nullptr);
    rmsnorm_kernel<<<TT, blk, 0, stream>>>(kva_f, KVA, KVLORA, kv_a_ln_w, kvc_bf, KVLORA);
    // 7) kv = kv_c @ kv_b_w
    gemm_bf16_wmma<<<grid(KVDIM), blk, 0, stream>>>(kvc_bf, KVLORA, kv_b_w, KVDIM, KVDIM, KVLORA, TT,
        nullptr, nullptr, 0, kv_f, KVDIM, nullptr, 0, nullptr, 0, nullptr, nullptr, nullptr);
    // 8) RoPE + assemble qf/kf (bf16, q pre-scaled by 1/sqrt(96))
    prep_qkv_kernel<<<TT, dim3(128), 0, stream>>>(q_f, kva_f, kv_f, positions, qf_bf, kf_bf);
    // 9) flash attention -> o bf16
    attn_flash_kernel<<<dim3(TT / 128, NH), blk, 0, stream>>>(qf_bf, kf_bf, kv_f, o_bf);
    // 10) attn_out = o @ o_w ; resid2 = x0 + attn_out
    gemm_bf16_wmma<<<grid(HD), blk, 0, stream>>>(o_bf, NH * VD, o_w, HD, HD, NH * VD, TT,
        nullptr, nullptr, 1, resid2, HD, nullptr, 0, x0, HD, nullptr, nullptr, nullptr);
    // 11) h2 = rmsnorm(resid2)
    rmsnorm_kernel<<<TT, blk, 0, stream>>>(resid2, HD, HD, post_ln_w, h2_bf, HD);
    // 12) gate + deterministic expert lists
    gate_kernel<<<TT, blk, 0, stream>>>(h2_bf, gate_w, gate_bias, sel, wts);
    build_lists_kernel<<<1, blk, 0, stream>>>(sel, wts, eidx, ew, ecnt);
    // 13) routed experts (gathered, top-2 only)
    hipMemsetAsync(routed, 0, (size_t)TT * HD * 4, stream);
    for (int e = 0; e < NEXP; e++) {
        const float* gw = e_gate_w + (size_t)e * HD * MOEI;
        const float* uw = e_up_w   + (size_t)e * HD * MOEI;
        const float* dw = e_down_w + (size_t)e * MOEI * HD;
        const int* gi = eidx + e * TT;
        const float* rs = ew + e * TT;
        const int* cp = ecnt + e;
        // up: u = gather(h2) @ uw
        gemm_bf16_wmma<<<grid(MOEI), blk, 0, stream>>>(h2_bf, HD, uw, MOEI, MOEI, HD, TT,
            gi, cp, 0, u_f, MOEI, nullptr, 0, nullptr, 0, nullptr, nullptr, nullptr);
        // gate: a = silu(gather(h2) @ gw) * u   (bf16)
        gemm_bf16_wmma<<<grid(MOEI), blk, 0, stream>>>(h2_bf, HD, gw, MOEI, MOEI, HD, TT,
            gi, cp, 3, nullptr, 0, a_bf, MOEI, u_f, MOEI, nullptr, nullptr, nullptr);
        // down: routed[token] += w * (a @ dw)
        gemm_bf16_wmma<<<grid(HD), blk, 0, stream>>>(a_bf, MOEI, dw, HD, HD, MOEI, TT,
            nullptr, cp, 4, routed, HD, nullptr, 0, nullptr, 0, nullptr, rs, gi);
    }
    // 14) shared expert, final: out = resid2 + routed + shared_down
    gemm_bf16_wmma<<<grid(MOEI), blk, 0, stream>>>(h2_bf, HD, s_up_w, MOEI, MOEI, HD, TT,
        nullptr, nullptr, 0, u_f, MOEI, nullptr, 0, nullptr, 0, nullptr, nullptr, nullptr);
    gemm_bf16_wmma<<<grid(MOEI), blk, 0, stream>>>(h2_bf, HD, s_gate_w, MOEI, MOEI, HD, TT,
        nullptr, nullptr, 3, nullptr, 0, a_bf, MOEI, u_f, MOEI, nullptr, nullptr, nullptr);
    gemm_bf16_wmma<<<grid(HD), blk, 0, stream>>>(a_bf, MOEI, s_down_w, HD, HD, MOEI, TT,
        nullptr, nullptr, 5, out, HD, nullptr, 0, resid2, HD, routed, nullptr, nullptr);
}
